// CausalSelfAttention_30365418783104
// MI455X (gfx1250) — matrix operand order repaired and checked by reading, not compiled, not run on hardware
//
#include <hip/hip_runtime.h>
#include <hip/hip_bf16.h>

// Problem constants (match reference: B=2, T=2048, C=1024, H=16, D=64)
#define BATCH   2
#define SEQ_T   2048
#define N_EMBD  1024
#define N_HEADS 16
#define HEAD_D  64
#define MROWS   (BATCH * SEQ_T)   // 4096

typedef __bf16 bf16;
typedef __attribute__((ext_vector_type(16))) __bf16 v16bf;
typedef __attribute__((ext_vector_type(8)))  float  v8f;
typedef __attribute__((ext_vector_type(4)))  unsigned int v4u;
typedef __attribute__((ext_vector_type(4)))  int v4i;

#ifndef __has_builtin
#define __has_builtin(x) 0
#endif
#if __has_builtin(__builtin_amdgcn_global_load_async_to_lds_b128) && \
    __has_builtin(__builtin_amdgcn_s_wait_asynccnt)
#define HAVE_ASYNC_LDS 1
#else
#define HAVE_ASYNC_LDS 0
#endif

union Frag32 { v4u u[2]; v16bf v; };

// Load a 16-element bf16 fragment as two contiguous 16B chunks.
__device__ __forceinline__ v16bf ld_frag(const bf16* p0, const bf16* p1) {
  Frag32 f;
  f.u[0] = *(const v4u*)p0;
  f.u[1] = *(const v4u*)p1;
  return f.v;
}

__device__ __forceinline__ v8f wmma_bf16(v16bf a, v16bf b, v8f c) {
  // D = A(16x32 bf16) * B(32x16 bf16) + C(16x16 f32)
  return __builtin_amdgcn_wmma_f32_16x16x32_bf16(
      /*neg_a=*/false, a, /*neg_b=*/false, b,
      /*c_mod=*/(short)0, c, /*reuse_a=*/false, /*reuse_b=*/false);
}

// 16B global -> LDS copy. Async (ASYNCcnt-tracked) when the toolchain exposes
// the gfx1250 builtin; plain load/store otherwise. The builtin takes
// (as1 v4i*, as3 v4i*, imm offset, imm cpol); pointers are produced via
// integer casts: low 32 bits of a flat shared address are the LDS byte offset
// (aperture rules), the as1 value equals the flat global address.
__device__ __forceinline__ void cp16_g2l(const bf16* g, bf16* l) {
#if HAVE_ASYNC_LDS
  __builtin_amdgcn_global_load_async_to_lds_b128(
      (__attribute__((address_space(1))) v4i*)(unsigned long long)(size_t)g,
      (__attribute__((address_space(3))) v4i*)(unsigned)(size_t)l,
      /*offset=*/0, /*cpol=*/0);
#else
  *(v4u*)l = *(const v4u*)g;
#endif
}

__device__ __forceinline__ void wait_async_prev_stage() {
#if HAVE_ASYNC_LDS
  // Two async-load instructions per stage per wave; <=2 outstanding means the
  // previous stage (loads complete in order) has fully landed in LDS.
  __builtin_amdgcn_s_wait_asynccnt(2);
#endif
}

// Stage a 64(N) x 32(K) bf16 tile of W (row-major [out,in]) into LDS,
// row-major [64][32]; 256 x 16B chunks spread across 128 threads.
__device__ __forceinline__ void stage_B(bf16* dst, const bf16* __restrict__ W,
                                        int n0, int k0, int tid) {
#pragma unroll
  for (int i = 0; i < 2; ++i) {
    const int ci  = tid + i * 128;      // chunk id 0..255
    const int row = ci >> 2;            // 4 chunks per 32-elt row
    const int col = (ci & 3) * 8;
    cp16_g2l(W + (size_t)(n0 + row) * N_EMBD + k0 + col, dst + row * 32 + col);
  }
}

// ---------------------------------------------------------------------------
// fp32 -> bf16 conversion, 8 elements per thread (b128 in/out).
// ---------------------------------------------------------------------------
__global__ __launch_bounds__(256) void f2bf_kernel(const float* __restrict__ in,
                                                   bf16* __restrict__ out, int n8) {
  const int i = blockIdx.x * 256 + threadIdx.x;
  if (i < n8) {
    const float4* p = (const float4*)in + (size_t)i * 2;
    const float4 x0 = p[0];
    const float4 x1 = p[1];
    union { v4u u; bf16 h[8]; } o;
    o.h[0] = (bf16)x0.x; o.h[1] = (bf16)x0.y; o.h[2] = (bf16)x0.z; o.h[3] = (bf16)x0.w;
    o.h[4] = (bf16)x1.x; o.h[5] = (bf16)x1.y; o.h[6] = (bf16)x1.z; o.h[7] = (bf16)x1.w;
    ((v4u*)out)[i] = o.u;
  }
}

// ---------------------------------------------------------------------------
// QKV projection: Out = X @ W^T  (X:[M,C] bf16, W:[C,C] bf16 row-major [out,in])
// Block = 128 threads = 4 waves; block tile 128(M) x 64(N); each wave owns
// 32(M) x 64(N) = 8 accumulators. W K-tiles double-buffered in LDS via async
// global->LDS copies; A fragments read directly (no cross-wave redundancy).
// Output scattered:
//   transposed==0 -> [B,H,T,D]  (Q, K;  Q pre-scaled by 1/sqrt(D))
//   transposed==1 -> [B,H,D,T]  (V, so attention PV B-fragments are contiguous)
// ---------------------------------------------------------------------------
__global__ __launch_bounds__(128) void qkv_gemm_kernel(
    const bf16* __restrict__ X, const bf16* __restrict__ W,
    bf16* __restrict__ Out, int transposed, float scale) {
  __shared__ bf16 sB[2][64 * 32];   // double-buffered W tile

  const int tid  = threadIdx.x;
  const int lane = tid & 31;
  const int wave = tid >> 5;
  const int lh   = lane >> 4;
  const int ln   = lane & 15;
  const int m0   = blockIdx.y * 128 + wave * 32;
  const int n0   = blockIdx.x * 64;

  v8f acc[2][4] = {};

  const bf16* arow0 = X + (size_t)(m0 + ln) * N_EMBD;
  const bf16* arow1 = X + (size_t)(m0 + 16 + ln) * N_EMBD;

  stage_B(sB[0], W, n0, 0, tid);
  for (int k0 = 0; k0 < N_EMBD; k0 += 32) {
    const int cur   = (k0 >> 5) & 1;
    const int knext = (k0 + 32 < N_EMBD) ? k0 + 32 : 0;   // dummy last prefetch
    stage_B(sB[cur ^ 1], W, n0, knext, tid);
    wait_async_prev_stage();
    __syncthreads();

    v16bf a0 = ld_frag(arow0 + k0 + lh * 16, arow0 + k0 + lh * 16 + 8);
    v16bf a1 = ld_frag(arow1 + k0 + lh * 16, arow1 + k0 + lh * 16 + 8);
#pragma unroll
    for (int t = 0; t < 4; ++t) {
      const bf16* brow = &sB[cur][(t * 16 + ln) * 32 + lh * 16];
      v16bf b = ld_frag(brow, brow + 8);
      acc[0][t] = wmma_bf16(a0, b, acc[0][t]);
      acc[1][t] = wmma_bf16(a1, b, acc[1][t]);
    }
    __syncthreads();   // everyone done reading sB[cur] before it is re-staged
  }

  // C/D layout: VGPR r, lane -> (M = r + 8*lh, N = ln)
#pragma unroll
  for (int g2 = 0; g2 < 2; ++g2) {
#pragma unroll
    for (int t = 0; t < 4; ++t) {
#pragma unroll
      for (int r = 0; r < 8; ++r) {
        const int gm = m0 + g2 * 16 + r + 8 * lh;
        const int gn = n0 + t * 16 + ln;
        const int bb = gm >> 11;            // / SEQ_T
        const int tt = gm & (SEQ_T - 1);
        const int hh = gn >> 6;             // / HEAD_D
        const int dd = gn & (HEAD_D - 1);
        const size_t idx = transposed
            ? ((size_t)(bb * N_HEADS + hh) * HEAD_D + dd) * SEQ_T + tt
            : ((size_t)(bb * N_HEADS + hh) * SEQ_T + tt) * HEAD_D + dd;
        Out[idx] = (bf16)(acc[g2][t][r] * scale);
      }
    }
  }
}

// ---------------------------------------------------------------------------
// Flash attention (causal). One wave per 16 query rows; 32-key tiles.
// Q,K: [B,H,T,D] bf16 (Q pre-scaled by 1/8). Vt: [B,H,D,T] bf16.
// Y:  [B,T,C] bf16 (attention output, fed to the output projection).
// ---------------------------------------------------------------------------
__global__ __launch_bounds__(128) void attn_kernel(
    const bf16* __restrict__ Q, const bf16* __restrict__ K,
    const bf16* __restrict__ Vt, bf16* __restrict__ Y) {
  __shared__ bf16 sP[4][16 * 32];   // per-wave P bounce buffer (row-major 16x32)

  const int lane = threadIdx.x & 31;
  const int wave = threadIdx.x >> 5;
  const int lh   = lane >> 4;
  const int ln   = lane & 15;
  const int q0   = (blockIdx.x * 4 + wave) * 16;
  const int bh   = blockIdx.y;
  const int b    = bh / N_HEADS;
  const int h    = bh % N_HEADS;

  // Q A-fragments for the two 32-wide K-dim chunks of D=64 (loaded once).
  const bf16* qrow = Q + ((size_t)bh * SEQ_T + q0 + ln) * HEAD_D;
  const v16bf qa0 = ld_frag(qrow + lh * 16,      qrow + lh * 16 + 8);
  const v16bf qa1 = ld_frag(qrow + 32 + lh * 16, qrow + 40 + lh * 16);

  const bf16* Kp = K  + (size_t)bh * SEQ_T * HEAD_D;
  const bf16* Vp = Vt + (size_t)bh * HEAD_D * SEQ_T;

  v8f o0 = {}, o1 = {}, o2 = {}, o3 = {};
  float ms[8], ls[8];
#pragma unroll
  for (int r = 0; r < 8; ++r) { ms[r] = -1e30f; ls[r] = 0.0f; }

  bf16* sp = sP[wave];
  const int jend = q0 + 16;                 // keys needed: <= q0+15
  for (int j = 0; j < jend; j += 32) {
    // --- S = Q @ K^T for 32 keys (two 16-col accumulators) ---
    const bf16* kr0 = Kp + (size_t)(j + ln) * HEAD_D;
    const bf16* kr1 = Kp + (size_t)(j + 16 + ln) * HEAD_D;
    v8f s0 = {}, s1 = {};
    s0 = wmma_bf16(qa0, ld_frag(kr0 + lh * 16,      kr0 + lh * 16 + 8),  s0);
    s0 = wmma_bf16(qa1, ld_frag(kr0 + 32 + lh * 16, kr0 + 40 + lh * 16), s0);
    s1 = wmma_bf16(qa0, ld_frag(kr1 + lh * 16,      kr1 + lh * 16 + 8),  s1);
    s1 = wmma_bf16(qa1, ld_frag(kr1 + 32 + lh * 16, kr1 + 40 + lh * 16), s1);

    // --- causal mask + online softmax (rows live on one 16-lane half) ---
    const int c0 = j + ln;
    const int c1 = j + 16 + ln;
    float alpha[8];
#pragma unroll
    for (int r = 0; r < 8; ++r) {
      const int row = q0 + 8 * lh + r;
      float a0 = (c0 <= row) ? s0[r] : -1e30f;
      float a1 = (c1 <= row) ? s1[r] : -1e30f;
      float mx = fmaxf(a0, a1);
      mx = fmaxf(mx, __shfl_xor(mx, 1, 32));
      mx = fmaxf(mx, __shfl_xor(mx, 2, 32));
      mx = fmaxf(mx, __shfl_xor(mx, 4, 32));
      mx = fmaxf(mx, __shfl_xor(mx, 8, 32));
      const float mnew = fmaxf(ms[r], mx);
      const float al = __expf(ms[r] - mnew);
      const float e0 = __expf(a0 - mnew);
      const float e1 = __expf(a1 - mnew);
      float rs = e0 + e1;
      rs += __shfl_xor(rs, 1, 32);
      rs += __shfl_xor(rs, 2, 32);
      rs += __shfl_xor(rs, 4, 32);
      rs += __shfl_xor(rs, 8, 32);
      ls[r] = ls[r] * al + rs;
      ms[r] = mnew;
      alpha[r] = al;
      // C-layout -> row-major LDS: row = r + 8*lh, col = ln / 16+ln
      sp[(8 * lh + r) * 32 + ln]      = (bf16)e0;
      sp[(8 * lh + r) * 32 + 16 + ln] = (bf16)e1;
    }

    // rescale running output
#pragma unroll
    for (int r = 0; r < 8; ++r) {
      o0[r] *= alpha[r]; o1[r] *= alpha[r]; o2[r] *= alpha[r]; o3[r] *= alpha[r];
    }

    // --- O += P(16x32) @ V(32x64); P A-fragment re-read from LDS in A layout ---
    // (DS ops from the same wave are in-order, so the reads see the stores.)
    const bf16* pr = sp + ln * 32;
    v16bf pa = ld_frag(pr + lh * 16, pr + lh * 16 + 8);
    const bf16* vr0 = Vp + (size_t)(0  + ln) * SEQ_T + j + lh * 16;
    const bf16* vr1 = Vp + (size_t)(16 + ln) * SEQ_T + j + lh * 16;
    const bf16* vr2 = Vp + (size_t)(32 + ln) * SEQ_T + j + lh * 16;
    const bf16* vr3 = Vp + (size_t)(48 + ln) * SEQ_T + j + lh * 16;
    o0 = wmma_bf16(pa, ld_frag(vr0, vr0 + 8), o0);
    o1 = wmma_bf16(pa, ld_frag(vr1, vr1 + 8), o1);
    o2 = wmma_bf16(pa, ld_frag(vr2, vr2 + 8), o2);
    o3 = wmma_bf16(pa, ld_frag(vr3, vr3 + 8), o3);
  }

  // --- finalize: O /= l, store to [B,T,C] bf16 ---
  float inv[8];
#pragma unroll
  for (int r = 0; r < 8; ++r) inv[r] = 1.0f / ls[r];
#pragma unroll
  for (int r = 0; r < 8; ++r) {
    const int trow = q0 + 8 * lh + r;
    const size_t base = ((size_t)b * SEQ_T + trow) * N_EMBD + h * HEAD_D + ln;
    Y[base +  0] = (bf16)(o0[r] * inv[r]);
    Y[base + 16] = (bf16)(o1[r] * inv[r]);
    Y[base + 32] = (bf16)(o2[r] * inv[r]);
    Y[base + 48] = (bf16)(o3[r] * inv[r]);
  }
}

// ---------------------------------------------------------------------------
// Output projection: out = Y @ Wp^T + bp, fp32 output. Same LDS-staged tiling.
// ---------------------------------------------------------------------------
__global__ __launch_bounds__(128) void proj_gemm_kernel(
    const bf16* __restrict__ X, const bf16* __restrict__ W,
    const float* __restrict__ bias, float* __restrict__ Out) {
  __shared__ bf16 sB[2][64 * 32];

  const int tid  = threadIdx.x;
  const int lane = tid & 31;
  const int wave = tid >> 5;
  const int lh   = lane >> 4;
  const int ln   = lane & 15;
  const int m0   = blockIdx.y * 128 + wave * 32;
  const int n0   = blockIdx.x * 64;

  v8f acc[2][4] = {};
  const bf16* arow0 = X + (size_t)(m0 + ln) * N_EMBD;
  const bf16* arow1 = X + (size_t)(m0 + 16 + ln) * N_EMBD;

  stage_B(sB[0], W, n0, 0, tid);
  for (int k0 = 0; k0 < N_EMBD; k0 += 32) {
    const int cur   = (k0 >> 5) & 1;
    const int knext = (k0 + 32 < N_EMBD) ? k0 + 32 : 0;
    stage_B(sB[cur ^ 1], W, n0, knext, tid);
    wait_async_prev_stage();
    __syncthreads();

    v16bf a0 = ld_frag(arow0 + k0 + lh * 16, arow0 + k0 + lh * 16 + 8);
    v16bf a1 = ld_frag(arow1 + k0 + lh * 16, arow1 + k0 + lh * 16 + 8);
#pragma unroll
    for (int t = 0; t < 4; ++t) {
      const bf16* brow = &sB[cur][(t * 16 + ln) * 32 + lh * 16];
      v16bf b = ld_frag(brow, brow + 8);
      acc[0][t] = wmma_bf16(a0, b, acc[0][t]);
      acc[1][t] = wmma_bf16(a1, b, acc[1][t]);
    }
    __syncthreads();
  }

#pragma unroll
  for (int g2 = 0; g2 < 2; ++g2) {
#pragma unroll
    for (int t = 0; t < 4; ++t) {
#pragma unroll
      for (int r = 0; r < 8; ++r) {
        const int gm = m0 + g2 * 16 + r + 8 * lh;
        const int gn = n0 + t * 16 + ln;
        Out[(size_t)gm * N_EMBD + gn] = acc[g2][t][r] + bias[gn];
      }
    }
  }
}

// ---------------------------------------------------------------------------
// Host-side launch
// ---------------------------------------------------------------------------
extern "C" void kernel_launch(void* const* d_in, const int* in_sizes, int n_in,
                              void* d_out, int out_size, void* d_ws, size_t ws_size,
                              hipStream_t stream) {
  (void)in_sizes; (void)n_in; (void)out_size; (void)ws_size;

  const float* x  = (const float*)d_in[0];
  const float* Wk = (const float*)d_in[1];
  const float* Wq = (const float*)d_in[2];
  const float* Wv = (const float*)d_in[3];
  const float* Wp = (const float*)d_in[4];
  const float* bp = (const float*)d_in[5];
  float* out = (float*)d_out;

  // Workspace carve-up (bf16 staging; ~48 MB total)
  char* ws = (char*)d_ws;
  size_t off = 0;
  auto carve = [&](size_t bytes) {
    void* p = ws + off;
    off += (bytes + 255) & ~(size_t)255;
    return p;
  };
  const size_t xmb = (size_t)MROWS * N_EMBD * sizeof(bf16);
  const size_t wmb = (size_t)N_EMBD * N_EMBD * sizeof(bf16);
  bf16* xb  = (bf16*)carve(xmb);
  bf16* Wkb = (bf16*)carve(wmb);
  bf16* Wqb = (bf16*)carve(wmb);
  bf16* Wvb = (bf16*)carve(wmb);
  bf16* Wpb = (bf16*)carve(wmb);
  bf16* Kh  = (bf16*)carve(xmb);   // [B,H,T,D]
  bf16* Qh  = (bf16*)carve(xmb);   // [B,H,T,D], pre-scaled by 1/8
  bf16* Vt  = (bf16*)carve(xmb);   // [B,H,D,T]
  bf16* Yw  = (bf16*)carve(xmb);   // [B,T,C]

  const int nx8 = MROWS * N_EMBD / 8;
  const int nw8 = N_EMBD * N_EMBD / 8;
  f2bf_kernel<<<dim3((nx8 + 255) / 256), dim3(256), 0, stream>>>(x,  xb,  nx8);
  f2bf_kernel<<<dim3((nw8 + 255) / 256), dim3(256), 0, stream>>>(Wk, Wkb, nw8);
  f2bf_kernel<<<dim3((nw8 + 255) / 256), dim3(256), 0, stream>>>(Wq, Wqb, nw8);
  f2bf_kernel<<<dim3((nw8 + 255) / 256), dim3(256), 0, stream>>>(Wv, Wvb, nw8);
  f2bf_kernel<<<dim3((nw8 + 255) / 256), dim3(256), 0, stream>>>(Wp, Wpb, nw8);

  const dim3 blk(128);
  const dim3 ggemm(N_EMBD / 64, MROWS / 128);
  qkv_gemm_kernel<<<ggemm, blk, 0, stream>>>(xb, Wkb, Kh, 0, 1.0f);
  qkv_gemm_kernel<<<ggemm, blk, 0, stream>>>(xb, Wqb, Qh, 0, 0.125f);  // 1/sqrt(64)
  qkv_gemm_kernel<<<ggemm, blk, 0, stream>>>(xb, Wvb, Vt, 1, 1.0f);

  const dim3 gattn(SEQ_T / 64, BATCH * N_HEADS);
  attn_kernel<<<gattn, blk, 0, stream>>>(Qh, Kh, Vt, Yw);

  proj_gemm_kernel<<<ggemm, blk, 0, stream>>>(Yw, Wpb, bp, out);
}
